// VoxelTransformerBloack_HardVoxel_29257317220734
// MI455X (gfx1250) — matrix operand order repaired and checked by reading, not compiled, not run on hardware
//
#include <hip/hip_runtime.h>
#include <hip/hip_bf16.h>

typedef __bf16 v16bf __attribute__((ext_vector_type(16)));
typedef __bf16 v8bf  __attribute__((ext_vector_type(8)));
typedef float  v8f   __attribute__((ext_vector_type(8)));

#define BN_EPS 1e-3f

// Build a v16bf fragment from two contiguous 16-byte LDS runs (ds_load_b128 x2)
__device__ __forceinline__ v16bf frag_from_runs(const __bf16* p0, const __bf16* p1) {
    v8bf lo = *(const v8bf*)p0;
    v8bf hi = *(const v8bf*)p1;
    return __builtin_shufflevector(lo, hi, 0, 1, 2, 3, 4, 5, 6, 7,
                                   8, 9, 10, 11, 12, 13, 14, 15);
}

// ---------------------------------------------------------------------------
// Tiled GEMM: C[M,N] = relu( alpha * (A @ B) + bias ), M,N multiples of 64.
// 256 threads / block, 64x64 output tile, bf16 WMMA with f32 accumulate.
// A: fp32 (M x K) row-major, or TA=1: A[k*lda + m]. B: fp32 (K x N) row-major.
// A/B staged in LDS (coalesced b32 global loads, bf16 convert); B stored
// transposed so both fragments are contiguous 16B runs -> ds_load_b128.
// ---------------------------------------------------------------------------
template <int TA>
__global__ __launch_bounds__(256) void gemm_tiled(
    const float* __restrict__ A, const float* __restrict__ B,
    const float* __restrict__ bias, float* __restrict__ C,
    int Mr, int Nc, int K, int lda, int ldb, int ldc,
    float alpha, int do_relu)
{
    constexpr int ASTR = 40;   // padded bf16 stride (80 B)
    constexpr int BSTR = 40;
    __shared__ __attribute__((aligned(16))) __bf16 As[64 * ASTR];   // 64 rows x 32 k
    __shared__ __attribute__((aligned(16))) __bf16 BsT[64 * BSTR];  // 64 cols x 32 k

    const int tid  = threadIdx.x;
    const int wid  = tid >> 5, lane = tid & 31;
    const int hi   = lane >> 4;
    const int l15  = lane & 15;
    const int row0 = blockIdx.x * 64, col0 = blockIdx.y * 64;

    v8f acc[2] = {v8f{}, v8f{}};

    for (int k0 = 0; k0 < K; k0 += 32) {
        // ---- stage A tile (64 x 32)
        if (TA == 0) {
            if (k0 + 32 < K)
                __builtin_prefetch(&A[(size_t)(row0 + (tid >> 5) * 8) * lda + k0 + 32], 0, 0);
            for (int idx = tid; idx < 2048; idx += 256) {
                int r = idx >> 5, c = idx & 31;
                int kk = k0 + c;
                float v = (kk < K) ? A[(size_t)(row0 + r) * lda + kk] : 0.f;
                As[r * ASTR + c] = (__bf16)v;
            }
        } else {
            for (int idx = tid; idx < 2048; idx += 256) {
                int r = idx & 63, c = idx >> 6;   // coalesced over r
                int kk = k0 + c;
                float v = (kk < K) ? A[(size_t)kk * lda + row0 + r] : 0.f;
                As[r * ASTR + c] = (__bf16)v;
            }
        }
        // ---- stage B tile (32 x 64), transposed into BsT[col][k]
        for (int idx = tid; idx < 2048; idx += 256) {
            int r = idx >> 6, c = idx & 63;       // coalesced over c
            int kk = k0 + r;
            float v = (kk < K) ? B[(size_t)kk * ldb + col0 + c] : 0.f;
            BsT[c * BSTR + r] = (__bf16)v;
        }
        __syncthreads();

        // ---- 2 subtiles per wave: tile ids wid and wid+8 (tm = t>>2, tn = t&3)
        #pragma unroll
        for (int tt = 0; tt < 2; ++tt) {
            int t = wid + tt * 8;
            int tm = t >> 2, tn = t & 3;
            const __bf16* ap = As + (tm * 16 + l15) * ASTR + hi * 8;
            const __bf16* bp = BsT + (tn * 16 + l15) * BSTR + hi * 8;
            v16bf av = frag_from_runs(ap, ap + 16);
            v16bf bv = frag_from_runs(bp, bp + 16);
            acc[tt] = __builtin_amdgcn_wmma_f32_16x16x32_bf16(false, av, false, bv,
                                                              (short)0, acc[tt], false, false);
        }
        __syncthreads();
    }

    #pragma unroll
    for (int tt = 0; tt < 2; ++tt) {
        int t = wid + tt * 8;
        int tm = t >> 2, tn = t & 3;
        int col = col0 + tn * 16 + l15;
        float bcol = bias ? bias[col] : 0.f;
        #pragma unroll
        for (int r = 0; r < 8; ++r) {
            int row = row0 + tm * 16 + r + hi * 8;
            float v = acc[tt][r] * alpha + bcol;
            if (do_relu) v = fmaxf(v, 0.f);
            C[(size_t)row * ldc + col] = v;
        }
    }
}

// ---------------------------------------------------------------------------
// Fused point attention: one block per (b, i). Three 64x64x64 WMMA GEMMs in
// LDS (fc_delta L2, fc_epsilon L1/L2), per-column softmax over j, keep only
// the diagonal (j == i) term. Weights staged transposed; all fragments are
// contiguous 16B LDS runs. fp32 softmax buffer aliases dead bf16 buffers.
// ---------------------------------------------------------------------------
#define PASTR 72   // padded bf16 stride (144 B, 16B-aligned rows)

__device__ __forceinline__ v8f wmma_tile_pa(const __bf16* __restrict__ As,
                                            const __bf16* __restrict__ WT,
                                            int tm, int tn, int lane)
{
    v8f acc = {};
    const int hi = lane >> 4;
    const int l15 = lane & 15;
    const __bf16* ap = As + (tm * 16 + l15) * PASTR;
    const __bf16* bp = WT + (tn * 16 + l15) * PASTR;
    #pragma unroll
    for (int k0 = 0; k0 < 64; k0 += 32) {
        v16bf av = frag_from_runs(ap + k0 + hi * 8, ap + k0 + 16 + hi * 8);
        v16bf bv = frag_from_runs(bp + k0 + hi * 8, bp + k0 + 16 + hi * 8);
        acc = __builtin_amdgcn_wmma_f32_16x16x32_bf16(false, av, false, bv,
                                                      (short)0, acc, false, false);
    }
    return acc;
}

__global__ __launch_bounds__(256) void point_attn_fused(
    const float* __restrict__ p_xyz,                       // (256,64,3)
    const float* __restrict__ qbuf, const float* __restrict__ kbuf,
    const float* __restrict__ vbuf,                        // (16384,64) each
    const float* __restrict__ Wd1, const float* __restrict__ bd1,  // 3x64
    const float* __restrict__ Wd2, const float* __restrict__ bd2,  // 64x64
    const float* __restrict__ We1, const float* __restrict__ be1,  // 64x64
    const float* __restrict__ We2, const float* __restrict__ be2,  // 64x64
    float* __restrict__ res)                               // (16384,64)
{
    const int b = blockIdx.x;
    const int i = blockIdx.y;
    const int tid = threadIdx.x;
    const int wid = tid >> 5, lane = tid & 31;
    const int hi = lane >> 4;
    const int l15 = lane & 15;
    const float scale = 0.125f; // 1/sqrt(64)

    // manually carved LDS pool (46,336 B); s_w (fp32) aliases dead s_A1/s_A2
    __shared__ __attribute__((aligned(16))) unsigned char s_pool[46336];
    float*  s_xyz  = (float*)(s_pool);                 //  768 B
    float*  s_q    = (float*)(s_pool + 768);           //  256 B
    float*  s_bias = (float*)(s_pool + 1024);          //  256 B
    __bf16* s_k    = (__bf16*)(s_pool + 1280);         // 8192 B, stride 64
    __bf16* s_A1   = (__bf16*)(s_pool + 9472);         // 9216 B (d1), stride 72
    __bf16* s_A2   = (__bf16*)(s_pool + 18688);        // 9216 B (t)
    __bf16* s_A3   = (__bf16*)(s_pool + 27904);        // 9216 B (h2)
    __bf16* s_WT   = (__bf16*)(s_pool + 37120);        // 9216 B (weight^T)
    float*  s_w    = (float*)(s_pool + 9472);          // 16384 B, aliases s_A1/s_A2

    const float* xyzB = p_xyz + (size_t)b * 64 * 3;
    for (int t = tid; t < 192; t += 256) s_xyz[t] = xyzB[t];
    if (tid < 64) s_q[tid] = qbuf[((size_t)b * 64 + i) * 64 + tid];
    for (int t = tid; t < 4096; t += 256)
        s_k[t] = (__bf16)kbuf[(size_t)b * 64 * 64 + t];
    for (int t = tid; t < 4096; t += 256)               // Wd2^T
        s_WT[(t & 63) * PASTR + (t >> 6)] = (__bf16)Wd2[t];
    if (tid < 64) s_bias[tid] = bd2[tid];
    __syncthreads();

    // fc_delta L1 (K=3, VALU): d1[j,h] = relu(|xyz_i - xyz_j| . Wd1[:,h] + bd1[h])
    {
        const int j = tid >> 2;
        const float xi0 = s_xyz[i * 3 + 0], xi1 = s_xyz[i * 3 + 1], xi2 = s_xyz[i * 3 + 2];
        const float d0 = fabsf(xi0 - s_xyz[j * 3 + 0]);
        const float d1 = fabsf(xi1 - s_xyz[j * 3 + 1]);
        const float d2 = fabsf(xi2 - s_xyz[j * 3 + 2]);
        #pragma unroll
        for (int hh = 0; hh < 16; ++hh) {
            int h = (tid & 3) * 16 + hh;
            float v = bd1[h] + d0 * Wd1[0 * 64 + h] + d1 * Wd1[1 * 64 + h] + d2 * Wd1[2 * 64 + h];
            s_A1[j * PASTR + h] = (__bf16)fmaxf(v, 0.f);
        }
    }
    __syncthreads();

    // GEMM1: t = (q_i - k_j + (d1 @ Wd2 + bd2)) * scale
    for (int t = wid; t < 16; t += 8) {
        int tm = t >> 2, tn = t & 3;
        v8f acc = wmma_tile_pa(s_A1, s_WT, tm, tn, lane);
        #pragma unroll
        for (int r = 0; r < 8; ++r) {
            int j = tm * 16 + r + hi * 8;
            int f = tn * 16 + l15;
            float tv = (s_q[f] - (float)s_k[j * 64 + f] + acc[r] + s_bias[f]) * scale;
            s_A2[j * PASTR + f] = (__bf16)tv;
        }
    }
    __syncthreads();
    for (int t = tid; t < 4096; t += 256)               // We1^T
        s_WT[(t & 63) * PASTR + (t >> 6)] = (__bf16)We1[t];
    if (tid < 64) s_bias[tid] = be1[tid];
    __syncthreads();

    // GEMM2: h2 = relu(t @ We1 + be1)
    for (int t = wid; t < 16; t += 8) {
        int tm = t >> 2, tn = t & 3;
        v8f acc = wmma_tile_pa(s_A2, s_WT, tm, tn, lane);
        #pragma unroll
        for (int r = 0; r < 8; ++r) {
            int j = tm * 16 + r + hi * 8;
            int f = tn * 16 + l15;
            s_A3[j * PASTR + f] = (__bf16)fmaxf(acc[r] + s_bias[f], 0.f);
        }
    }
    __syncthreads();
    for (int t = tid; t < 4096; t += 256)               // We2^T
        s_WT[(t & 63) * PASTR + (t >> 6)] = (__bf16)We2[t];
    if (tid < 64) s_bias[tid] = be2[tid];
    __syncthreads();

    // GEMM3: w = h2 @ We2 + be2  (fp32 into aliased buffer for the softmax)
    for (int t = wid; t < 16; t += 8) {
        int tm = t >> 2, tn = t & 3;
        v8f acc = wmma_tile_pa(s_A3, s_WT, tm, tn, lane);
        #pragma unroll
        for (int r = 0; r < 8; ++r) {
            int j = tm * 16 + r + hi * 8;
            int f = tn * 16 + l15;
            s_w[j * 64 + f] = acc[r] + s_bias[f];
        }
    }
    __syncthreads();

    // softmax over j per column f; keep diagonal j == i, scale by v[b,i,f]
    if (tid < 64) {
        const int f = tid;
        float mx = -1e30f;
        for (int j = 0; j < 64; ++j) mx = fmaxf(mx, s_w[j * 64 + f]);
        float sum = 0.f;
        for (int j = 0; j < 64; ++j) sum += __expf(s_w[j * 64 + f] - mx);
        float diag = __expf(s_w[i * 64 + f] - mx) / sum;
        float vv = vbuf[((size_t)b * 64 + i) * 64 + f];
        res[((size_t)b * 64 + i) * 64 + f] = diag * vv;
    }
}

// ---------------------------------------------------------------------------
// BatchNorm batch statistics: one block per column
// ---------------------------------------------------------------------------
__global__ __launch_bounds__(256) void col_stats(const float* __restrict__ X,
                                                 int R, int ld,
                                                 float* __restrict__ mean,
                                                 float* __restrict__ var)
{
    const int c = blockIdx.x;
    __shared__ float ssum[256], ssq[256];
    float s = 0.f, q = 0.f;
    for (int r = threadIdx.x; r < R; r += 256) {
        float v = X[(size_t)r * ld + c];
        s += v; q += v * v;
    }
    ssum[threadIdx.x] = s; ssq[threadIdx.x] = q;
    __syncthreads();
    for (int st = 128; st > 0; st >>= 1) {
        if (threadIdx.x < st) {
            ssum[threadIdx.x] += ssum[threadIdx.x + st];
            ssq[threadIdx.x]  += ssq[threadIdx.x + st];
        }
        __syncthreads();
    }
    if (threadIdx.x == 0) {
        float m = ssum[0] / R;
        mean[c] = m;
        var[c] = ssq[0] / R - m * m;
    }
}

__global__ void bn_relu(const float* __restrict__ X, float* __restrict__ Y,
                        int R, int C, int ldx, int ldy,
                        const float* __restrict__ mean, const float* __restrict__ var,
                        const float* __restrict__ gamma, const float* __restrict__ beta)
{
    int idx = blockIdx.x * blockDim.x + threadIdx.x;
    if (idx >= R * C) return;
    int r = idx / C, c = idx % C;
    float v = X[(size_t)r * ldx + c];
    v = gamma[c] * (v - mean[c]) * rsqrtf(var[c] + BN_EPS) + beta[c];
    Y[(size_t)r * ldy + c] = fmaxf(v, 0.f);
}

__global__ void copy_cols(const float* __restrict__ src, float* __restrict__ dst,
                          int R, int C, int lds_, int ldd)
{
    int idx = blockIdx.x * blockDim.x + threadIdx.x;
    if (idx >= R * C) return;
    int r = idx / C, c = idx % C;
    dst[(size_t)r * ldd + c] = src[(size_t)r * lds_ + c];
}

// ---------------------------------------------------------------------------
// Furthest point sampling: serial 512-step scan, block-wide argmax.
// Tie-break: lowest index (jnp.argmax convention).
// ---------------------------------------------------------------------------
__global__ __launch_bounds__(1024) void fps_kernel(const float* __restrict__ xyz,
                                                   float* __restrict__ dist,
                                                   int* __restrict__ idx,
                                                   int P, int Ksel)
{
    __shared__ float sval[1024];
    __shared__ int   sidx[1024];
    __shared__ int   s_last;
    const int tid = threadIdx.x;
    for (int p = tid; p < P; p += 1024) dist[p] = 1e10f;
    if (tid == 0) s_last = 0;
    __syncthreads();
    for (int it = 0; it < Ksel; ++it) {
        const int last = s_last;
        if (tid == 0) idx[it] = last;
        const float lx = xyz[last * 3 + 0], ly = xyz[last * 3 + 1], lz = xyz[last * 3 + 2];
        float bm = -1.f; int bi = P;
        for (int p = tid; p < P; p += 1024) {
            float dx = xyz[p * 3 + 0] - lx;
            float dy = xyz[p * 3 + 1] - ly;
            float dz = xyz[p * 3 + 2] - lz;
            float nd = fminf(dist[p], dx * dx + dy * dy + dz * dz);
            dist[p] = nd;
            if (nd > bm || (nd == bm && p < bi)) { bm = nd; bi = p; }
        }
        sval[tid] = bm; sidx[tid] = bi;
        __syncthreads();
        for (int st = 512; st > 0; st >>= 1) {
            if (tid < st) {
                float ov = sval[tid + st]; int oi = sidx[tid + st];
                if (ov > sval[tid] || (ov == sval[tid] && oi < sidx[tid])) {
                    sval[tid] = ov; sidx[tid] = oi;
                }
            }
            __syncthreads();
        }
        if (tid == 0) s_last = sidx[0];
        __syncthreads();
    }
}

__global__ void gather_rows(const float* __restrict__ src, const int* __restrict__ idx,
                            float* __restrict__ dst, int C)
{
    int s = blockIdx.x, c = threadIdx.x;
    if (c < C) dst[(size_t)s * C + c] = src[(size_t)idx[s] * C + c];
}

__global__ __launch_bounds__(256) void max_over_m(const float* __restrict__ x2,
                                                  float* __restrict__ out, int ld)
{
    int b = blockIdx.x, c = threadIdx.x;
    float m = -1e30f;
    for (int mm = 0; mm < 64; ++mm)
        m = fmaxf(m, x2[((size_t)b * 64 + mm) * ld + c]);
    out[(size_t)b * 256 + c] = m;
}

// voxel attention collapses to sum over m (softmax over a size-1 axis == 1)
__global__ __launch_bounds__(256) void sum_over_m_add(const float* __restrict__ vatt,
                                                      const float* __restrict__ vfeat,
                                                      float* __restrict__ out)
{
    int b = blockIdx.x, c = threadIdx.x;
    float s = 0.f;
    for (int mm = 0; mm < 64; ++mm)
        s += vatt[((size_t)b * 64 + mm) * 256 + c];
    out[(size_t)b * 256 + c] = s + vfeat[(size_t)b * 256 + c];
}

// ---------------------------------------------------------------------------
// Host-side orchestration
// ---------------------------------------------------------------------------
extern "C" void kernel_launch(void* const* d_in, const int* in_sizes, int n_in,
                              void* d_out, int out_size, void* d_ws, size_t ws_size,
                              hipStream_t stream) {
    (void)in_sizes; (void)n_in; (void)out_size; (void)ws_size;
    const int N = 256, M = 64, NM = N * M;    // 16384
    (void)M;

    // ---- inputs (recursive dict-insertion-order flattening of setup_inputs)
    const float* p_xyz   = (const float*)d_in[0];
    const float* pcoors  = (const float*)d_in[3];
    const float* feats   = (const float*)d_in[2];
    const float* fc1_w = (const float*)d_in[5],  *fc1_b = (const float*)d_in[6];
    const float* fc3_w = (const float*)d_in[7],  *fc3_b = (const float*)d_in[8];
    const float* bn1_g = (const float*)d_in[9],  *bn1_b = (const float*)d_in[10];
    const float* bnp_g = (const float*)d_in[11], *bnp_b = (const float*)d_in[12];
    const float* bng_g = (const float*)d_in[13], *bng_b = (const float*)d_in[14];
    const float* bnv_g = (const float*)d_in[15], *bnv_b = (const float*)d_in[16];
    const float* bn2_g = (const float*)d_in[17], *bn2_b = (const float*)d_in[18];
    const float* pa_a1w = (const float*)d_in[19], *pa_a1b = (const float*)d_in[20];
    const float* pa_a2w = (const float*)d_in[21], *pa_a2b = (const float*)d_in[22];
    const float* pa_b1w = (const float*)d_in[23], *pa_b1b = (const float*)d_in[24];
    const float* pa_b2w = (const float*)d_in[25], *pa_b2b = (const float*)d_in[26];
    const float* pa_d1w = (const float*)d_in[27], *pa_d1b = (const float*)d_in[28];
    const float* pa_d2w = (const float*)d_in[29], *pa_d2b = (const float*)d_in[30];
    const float* pa_e1w = (const float*)d_in[31], *pa_e1b = (const float*)d_in[32];
    const float* pa_e2w = (const float*)d_in[33], *pa_e2b = (const float*)d_in[34];
    const float* pa_o1w = (const float*)d_in[35], *pa_o1b = (const float*)d_in[36];
    const float* pa_o2w = (const float*)d_in[37], *pa_o2b = (const float*)d_in[38];
    const float* ga_a1w = (const float*)d_in[39], *ga_a1b = (const float*)d_in[40];
    const float* ga_a2w = (const float*)d_in[41], *ga_a2b = (const float*)d_in[42];
    const float* ga_b1w = (const float*)d_in[43], *ga_b1b = (const float*)d_in[44];
    const float* ga_b2w = (const float*)d_in[45], *ga_b2b = (const float*)d_in[46];
    const float* ga_o1w = (const float*)d_in[47], *ga_o1b = (const float*)d_in[48];
    const float* ga_o2w = (const float*)d_in[49], *ga_o2b = (const float*)d_in[50];
    // voxelattn alpha/beta/delta are dead code (softmax over size-1 axis); only omega
    const float* va_o1w = (const float*)d_in[63], *va_o1b = (const float*)d_in[64];
    const float* va_o2w = (const float*)d_in[65], *va_o2b = (const float*)d_in[66];

    // ---- workspace layout (float units)
    float* ws = (float*)d_ws;
    const size_t U = 1u << 20;           // 1048576 = 16384*64
    float* X    = ws + 0 * U;            // (16384,64) post fc1-bn-relu
    float* H    = ws + 1 * U;            // MLP hidden temp
    float* QPA  = ws + 2 * U;
    float* KPA  = ws + 3 * U;
    float* VPA  = ws + 4 * U;
    float* RES  = ws + 5 * U;
    float* X2   = ws + 6 * U;            // (16384,256)
    float* QG   = ws + 10 * U;
    float* HV   = ws + 11 * U;           // (16384,256)
    float* VATT = ws + 15 * U;           // (16384,256)
    float* FSEL = ws + 19 * U;           // (512,64)
    float* HG   = FSEL + 32768;
    float* KG   = HG + 32768;
    float* VG   = KG + 32768;
    float* TMAT = VG + 32768;            // (64,64)
    float* VFEAT= TMAT + 4096;           // (256,256)
    float* RESV = VFEAT + 65536;         // (256,256)
    float* Y3   = RESV + 65536;          // (256,128)
    float* DIST = Y3 + 32768;            // (16384)
    float* MEAN = DIST + 16384;          // (256)
    float* VAR  = MEAN + 256;            // (256)
    int*   IDX  = (int*)(VAR + 256);     // (512)

    auto gemmN = [&](const float* A, const float* B, const float* bias, float* C,
                     int Mr, int Nc, int K, int lda, int ldb, int ldc,
                     float alpha, int relu) {
        dim3 g(Mr / 64, Nc / 64);
        gemm_tiled<0><<<g, 256, 0, stream>>>(A, B, bias, C, Mr, Nc, K, lda, ldb, ldc,
                                             alpha, relu);
    };
    auto gemmT = [&](const float* A, const float* B, const float* bias, float* C,
                     int Mr, int Nc, int K, int lda, int ldb, int ldc,
                     float alpha, int relu) {
        dim3 g(Mr / 64, Nc / 64);
        gemm_tiled<1><<<g, 256, 0, stream>>>(A, B, bias, C, Mr, Nc, K, lda, ldb, ldc,
                                             alpha, relu);
    };
    auto ew = [](int n) { return dim3((n + 255) / 256); };

    // 1) x = relu(bn1(features @ fc1))
    gemmN(feats, fc1_w, fc1_b, X, NM, 64, 16, 16, 64, 64, 1.f, 0);
    col_stats<<<64, 256, 0, stream>>>(X, NM, 64, MEAN, VAR);
    bn_relu<<<ew(NM * 64), 256, 0, stream>>>(X, X, NM, 64, 64, 64, MEAN, VAR, bn1_g, bn1_b);

    // 2) point attention q/k/v MLPs
    gemmN(p_xyz, pa_a1w, pa_a1b, H, NM, 64, 3, 3, 64, 64, 1.f, 1);
    gemmN(H, pa_a2w, pa_a2b, QPA, NM, 64, 64, 64, 64, 64, 1.f, 0);
    gemmN(p_xyz, pa_b1w, pa_b1b, H, NM, 64, 3, 3, 64, 64, 1.f, 1);
    gemmN(H, pa_b2w, pa_b2b, KPA, NM, 64, 64, 64, 64, 64, 1.f, 0);
    gemmN(X, pa_o1w, pa_o1b, H, NM, 64, 64, 64, 64, 64, 1.f, 1);
    gemmN(H, pa_o2w, pa_o2b, VPA, NM, 64, 64, 64, 64, 64, 1.f, 0);

    // 3) fused point attention (diag-softmax)
    point_attn_fused<<<dim3(N, 64), 256, 0, stream>>>(p_xyz, QPA, KPA, VPA,
        pa_d1w, pa_d1b, pa_d2w, pa_d2b, pa_e1w, pa_e1b, pa_e2w, pa_e2b, RES);

    // 4) local_f = relu(bnp(concat[res, x])) -> X2[:, 0:128]
    copy_cols<<<ew(NM * 64), 256, 0, stream>>>(RES, X2, NM, 64, 64, 256);
    copy_cols<<<ew(NM * 64), 256, 0, stream>>>(X, X2 + 64, NM, 64, 64, 256);
    col_stats<<<128, 256, 0, stream>>>(X2, NM, 256, MEAN, VAR);
    bn_relu<<<ew(NM * 128), 256, 0, stream>>>(X2, X2, NM, 128, 256, 256, MEAN, VAR, bnp_g, bnp_b);

    // 5) global attention: res = q @ (k^T v) / sqrt(64)  (no softmax -> associativity)
    gemmN(X, ga_a1w, ga_a1b, H, NM, 64, 64, 64, 64, 64, 1.f, 1);
    gemmN(H, ga_a2w, ga_a2b, QG, NM, 64, 64, 64, 64, 64, 1.f, 0);
    fps_kernel<<<1, 1024, 0, stream>>>(pcoors, DIST, IDX, NM, 512);
    gather_rows<<<512, 64, 0, stream>>>(X, IDX, FSEL, 64);
    gemmN(FSEL, ga_b1w, ga_b1b, HG, 512, 64, 64, 64, 64, 64, 1.f, 1);
    gemmN(HG, ga_b2w, ga_b2b, KG, 512, 64, 64, 64, 64, 64, 1.f, 0);
    gemmN(FSEL, ga_o1w, ga_o1b, HG, 512, 64, 64, 64, 64, 64, 1.f, 1);
    gemmN(HG, ga_o2w, ga_o2b, VG, 512, 64, 64, 64, 64, 64, 1.f, 0);
    gemmT(KG, VG, nullptr, TMAT, 64, 64, 512, 64, 64, 64, 1.f, 0);      // k^T @ v
    gemmN(QG, TMAT, nullptr, X2 + 128, NM, 64, 64, 64, 64, 256, 0.125f, 0);
    copy_cols<<<ew(NM * 64), 256, 0, stream>>>(X, X2 + 192, NM, 64, 64, 256);
    col_stats<<<128, 256, 0, stream>>>(X2 + 128, NM, 256, MEAN, VAR);
    bn_relu<<<ew(NM * 128), 256, 0, stream>>>(X2 + 128, X2 + 128, NM, 128, 256, 256,
                                              MEAN, VAR, bng_g, bng_b);

    // 6) v_feat = max over M ; voxel attention = sum_m fc_omega(x2) + v_feat
    max_over_m<<<N, 256, 0, stream>>>(X2, VFEAT, 256);
    gemmN(X2, va_o1w, va_o1b, HV, NM, 256, 256, 256, 256, 256, 1.f, 1);
    gemmN(HV, va_o2w, va_o2b, VATT, NM, 256, 256, 256, 256, 256, 1.f, 0);
    sum_over_m_add<<<N, 256, 0, stream>>>(VATT, VFEAT, RESV);
    col_stats<<<256, 256, 0, stream>>>(RESV, N, 256, MEAN, VAR);
    bn_relu<<<ew(N * 256), 256, 0, stream>>>(RESV, RESV, N, 256, 256, 256, MEAN, VAR, bnv_g, bnv_b);

    // 7) out = relu(bn2(fc3(voxel_f)))
    gemmN(RESV, fc3_w, fc3_b, Y3, N, 128, 256, 256, 128, 128, 1.f, 0);
    col_stats<<<128, 256, 0, stream>>>(Y3, N, 128, MEAN, VAR);
    bn_relu<<<ew(N * 128), 256, 0, stream>>>(Y3, (float*)d_out, N, 128, 128, 128,
                                             MEAN, VAR, bn2_g, bn2_b);
}
